// NearTransEncoder_27650999451843
// MI455X (gfx1250) — matrix operand order repaired and checked by reading, not compiled, not run on hardware
//
#include <hip/hip_runtime.h>
#include <hip/hip_bf16.h>
#include <stdint.h>

#define DMODEL 512
#define DINNER 2048
#define NHEAD  8
#define DKH    64
#define NLAYERS 4
#define BATCH  64
#define SMAX   12
#define REPIDX 1

typedef __attribute__((ext_vector_type(16))) __bf16 v16bf;
typedef __attribute__((ext_vector_type(8)))  float v8f;

union FragBF { v16bf v; uint4 q[2]; unsigned short h[16]; };
union U4H    { uint4 q; unsigned short h[8]; };

__device__ __forceinline__ unsigned short f2bf(float f) {
  unsigned int u = __float_as_uint(f);
  u += 0x7FFFu + ((u >> 16) & 1u);          // round-to-nearest-even
  return (unsigned short)(u >> 16);
}

__device__ __forceinline__ v8f wmma_bf16(const FragBF& a, const FragBF& b, v8f c) {
  return __builtin_amdgcn_wmma_f32_16x16x32_bf16(false, a.v, false, b.v, (short)0, c, false, false);
}

// Async global->LDS DMA (ASYNCcnt-tracked); lds_addr = wave-relative LDS byte address.
__device__ __forceinline__ void async_copy_b128(unsigned lds_addr, const void* gaddr) {
  asm volatile("global_load_async_to_lds_b128 %0, %1, off"
               :: "v"(lds_addr), "v"(gaddr) : "memory");
}
__device__ __forceinline__ void wait_async0() {
  asm volatile("s_wait_asynccnt 0x0" ::: "memory");
}

// ---- wave32 16-lane butterfly reductions in pure VALU (DPP), no LDS traffic.
// ctrl: 0xB1 = quad_perm xor1, 0x4E = quad_perm xor2,
//       0x141 = row_half_mirror (xor7 in 8-group), 0x140 = row_mirror (xor15).
template <int CTRL>
__device__ __forceinline__ float dppf(float v) {
  return __int_as_float(__builtin_amdgcn_update_dpp(0, __float_as_int(v), CTRL, 0xF, 0xF, true));
}
__device__ __forceinline__ float redmax16(float v) {
  v = fmaxf(v, dppf<0xB1>(v));
  v = fmaxf(v, dppf<0x4E>(v));
  v = fmaxf(v, dppf<0x141>(v));
  v = fmaxf(v, dppf<0x140>(v));
  return v;
}
__device__ __forceinline__ float redsum16(float v) {
  v += dppf<0xB1>(v);
  v += dppf<0x4E>(v);
  v += dppf<0x141>(v);
  v += dppf<0x140>(v);
  return v;
}

// ---------------------------------------------------------------- utilities
__global__ void cvt_bf16_kernel(const float* __restrict__ src,
                                unsigned short* __restrict__ dst, int n) {
  for (int i = blockIdx.x * blockDim.x + threadIdx.x; i < n; i += gridDim.x * blockDim.x)
    dst[i] = f2bf(src[i]);
}

// One thread per batch: serial ragged pack (32 POIs x <=12 tokens, cheap).
__global__ void pack_kernel(const int* __restrict__ npl, const int* __restrict__ ttn,
                            int* __restrict__ poi_idx, int* __restrict__ tok_idx,
                            float* __restrict__ validf, int L) {
  int b = blockIdx.x * blockDim.x + threadIdx.x;
  if (b >= BATCH) return;
  int start = 0;
  for (int i = 0; i < b; ++i) start += npl[i];
  int cnt = npl[b];
  for (int s = 0; s < L; ++s) { poi_idx[b * L + s] = 0; tok_idx[b * L + s] = 0; }
  int pos = 1;                                   // slot 0 = [REP]
  for (int p = start; p < start + cnt; ++p) {
    int t = ttn[p];
    for (int j = 0; j < t && pos < L; ++j) {
      poi_idx[b * L + pos] = p; tok_idx[b * L + pos] = j; ++pos;
    }
  }
  for (int s = 0; s < L; ++s) validf[b * L + s] = (s < pos) ? 1.0f : 0.0f;
}

// Embedding + sinusoid PE + loc broadcast, [REP] at slot0, then LayerNorm(lnf).
__global__ __launch_bounds__(256)
void embed_ln_kernel(const int* __restrict__ poi_type, const float* __restrict__ loc_emb,
                     const float* __restrict__ emb,
                     const int* __restrict__ poi_idx, const int* __restrict__ tok_idx,
                     const float* __restrict__ validf,
                     const float* __restrict__ lnf_g, const float* __restrict__ lnf_b,
                     float* __restrict__ x, unsigned short* __restrict__ xb, int L) {
  __shared__ float r1[256], r2[256];
  const int row = blockIdx.x;
  const int s = row % L;
  const int tid = threadIdx.x;
  const int p = poi_idx[row], t = tok_idx[row];
  const float msk = validf[row];
  float v[2];
#pragma unroll
  for (int e = 0; e < 2; ++e) {
    const int d = tid + e * 256;
    float val;
    if (s == 0) {
      val = emb[REPIDX * DMODEL + d];
    } else if (msk > 0.5f) {
      float den = __powf(10000.0f, (float)(2 * (d / 2)) * (1.0f / (float)DMODEL));
      float ang = (float)t / den;
      float pe  = (d & 1) ? __cosf(ang) : __sinf(ang);
      val = emb[poi_type[p * SMAX + t] * DMODEL + d] + pe + loc_emb[p * DMODEL + d];
    } else {
      val = 0.0f;
    }
    v[e] = val;
  }
  r1[tid] = v[0] + v[1];
  r2[tid] = v[0] * v[0] + v[1] * v[1];
  __syncthreads();
  for (int sft = 128; sft > 0; sft >>= 1) {
    if (tid < sft) { r1[tid] += r1[tid + sft]; r2[tid] += r2[tid + sft]; }
    __syncthreads();
  }
  const float mean = r1[0] * (1.0f / DMODEL);
  const float var  = r2[0] * (1.0f / DMODEL) - mean * mean;
  const float rstd = rsqrtf(var + 1e-6f);
#pragma unroll
  for (int e = 0; e < 2; ++e) {
    const int d = tid + e * 256;
    float o = lnf_g[d] * (v[e] - mean) * rstd + lnf_b[d];
    x[(size_t)row * DMODEL + d] = o;
    xb[(size_t)row * DMODEL + d] = f2bf(o);
  }
}

// Residual add + LayerNorm, writes f32 residual stream and bf16 matmul copy.
__global__ __launch_bounds__(256)
void resln_kernel(float* __restrict__ x, const float* __restrict__ t,
                  const float* __restrict__ g, const float* __restrict__ bta,
                  unsigned short* __restrict__ xb) {
  __shared__ float r1[256], r2[256];
  const int row = blockIdx.x;
  const int tid = threadIdx.x;
  float v0 = x[(size_t)row * DMODEL + tid]       + t[(size_t)row * DMODEL + tid];
  float v1 = x[(size_t)row * DMODEL + tid + 256] + t[(size_t)row * DMODEL + tid + 256];
  r1[tid] = v0 + v1;
  r2[tid] = v0 * v0 + v1 * v1;
  __syncthreads();
  for (int sft = 128; sft > 0; sft >>= 1) {
    if (tid < sft) { r1[tid] += r1[tid + sft]; r2[tid] += r2[tid + sft]; }
    __syncthreads();
  }
  const float mean = r1[0] * (1.0f / DMODEL);
  const float var  = r2[0] * (1.0f / DMODEL) - mean * mean;
  const float rstd = rsqrtf(var + 1e-6f);
  float o0 = g[tid] * (v0 - mean) * rstd + bta[tid];
  float o1 = g[tid + 256] * (v1 - mean) * rstd + bta[tid + 256];
  x[(size_t)row * DMODEL + tid]        = o0;
  x[(size_t)row * DMODEL + tid + 256]  = o1;
  xb[(size_t)row * DMODEL + tid]       = f2bf(o0);
  xb[(size_t)row * DMODEL + tid + 256] = f2bf(o1);
}

// ------------------------------------------------------------- WMMA GEMM
// C[M,N] = A[M,K](bf16, row-major) x B[K,N](bf16, row-major), f32 accumulate.
// Workgroup tile 128x64, 8 waves (4 along M, 2 along N), each wave 32x32.
// BK=64: 8 WMMA per staged tile between barriers; A-tile staged with async
// global->LDS DMA, next tile prefetched.
template <int OUTBF16, int RELU, int HASBIAS>
__global__ __launch_bounds__(256)
void gemm_kernel(const unsigned short* __restrict__ A, const unsigned short* __restrict__ Bw,
                 const float* __restrict__ bias,
                 float* __restrict__ Cf, unsigned short* __restrict__ Cb,
                 int M, int N, int K) {
  __shared__ unsigned short As[128 * 72];   // [row][k], ld=72 keeps 16B-aligned b128 loads
  __shared__ unsigned short Bt[64 * 72];    // transposed: [n][k]
  const int tid  = threadIdx.x;
  const int lane = tid & 31, lr = lane & 15, hi = lane >> 4;
  const int w = tid >> 5, wm = w & 3, wn = w >> 2;
  const int m0 = blockIdx.y * 128, n0 = blockIdx.x * 64;

  v8f acc[2][2] = {{{}, {}}, {{}, {}}};

  for (int kt = 0; kt < K; kt += 64) {
    // stage A via async DMA: 128 rows x 64 halfs (8 x 16B per row), 4 chunks/thread.
    // Out-of-range rows are skipped: they only feed masked output rows.
#pragma unroll
    for (int p = 0; p < 4; ++p) {
      const int idx = tid + p * 256;
      const int row = idx >> 3, c = (idx & 7) * 8;
      if (m0 + row < M) {
        async_copy_b128((unsigned)(size_t)(As + row * 72 + c),
                        A + (size_t)(m0 + row) * K + kt + c);
      }
    }
    // stage B transposed: 64 k-rows x 64 n-cols, 2 chunks/thread
#pragma unroll
    for (int p = 0; p < 2; ++p) {
      const int idx = tid + p * 256;
      const int r = idx >> 3, c = (idx & 7) * 8;
      U4H d; d.q = *(const uint4*)(Bw + (size_t)(kt + r) * N + n0 + c);
#pragma unroll
      for (int u = 0; u < 8; ++u) Bt[(c + u) * 72 + r] = d.h[u];
    }
    // speculative prefetch of the next K-tile (global_prefetch_b8)
    if (kt + 64 < K) {
      __builtin_prefetch(A + (size_t)(m0 + (tid >> 1)) * K + kt + 64 + (tid & 1) * 32, 0, 0);
      __builtin_prefetch(Bw + (size_t)(kt + 64 + (tid >> 2)) * N + n0 + (tid & 3) * 16, 0, 0);
    }
    wait_async0();                 // A DMA landed in LDS
    __syncthreads();

#pragma unroll
    for (int kk = 0; kk < 64; kk += 32) {
      FragBF a0, a1, b0, b1;
      const unsigned short* ar0 = As + (wm * 32 + lr) * 72 + kk;
      const unsigned short* ar1 = As + (wm * 32 + 16 + lr) * 72 + kk;
      a0.q[0] = *(const uint4*)(ar0 + hi * 16);  a0.q[1] = *(const uint4*)(ar0 + hi * 16 + 8);
      a1.q[0] = *(const uint4*)(ar1 + hi * 16);  a1.q[1] = *(const uint4*)(ar1 + hi * 16 + 8);
      const unsigned short* bc0 = Bt + (wn * 32 + lr) * 72 + kk;
      const unsigned short* bc1 = Bt + (wn * 32 + 16 + lr) * 72 + kk;
      b0.q[0] = *(const uint4*)(bc0 + hi * 16); b0.q[1] = *(const uint4*)(bc0 + hi * 16 + 8);
      b1.q[0] = *(const uint4*)(bc1 + hi * 16); b1.q[1] = *(const uint4*)(bc1 + hi * 16 + 8);

      acc[0][0] = wmma_bf16(a0, b0, acc[0][0]);
      acc[0][1] = wmma_bf16(a0, b1, acc[0][1]);
      acc[1][0] = wmma_bf16(a1, b0, acc[1][0]);
      acc[1][1] = wmma_bf16(a1, b1, acc[1][1]);
    }
    __syncthreads();
  }

#pragma unroll
  for (int i = 0; i < 2; ++i) {
#pragma unroll
    for (int j = 0; j < 2; ++j) {
#pragma unroll
      for (int r = 0; r < 8; ++r) {
        const int Mi = m0 + wm * 32 + i * 16 + r + hi * 8;
        const int Ni = n0 + wn * 32 + j * 16 + lr;
        if (Mi < M) {
          float v = acc[i][j][r];
          if (HASBIAS) v += bias[Ni];
          if (RELU)    v = fmaxf(v, 0.0f);
          if (OUTBF16) Cb[(size_t)Mi * N + Ni] = f2bf(v);
          else         Cf[(size_t)Mi * N + Ni] = v;
        }
      }
    }
  }
}

// ---------------------------------------------------------- flash attention
// One workgroup = 4 waves = 4 query tiles of 16 rows for one (batch, head).
// Keys streamed in LDS chunks of 128 (K via async DMA); online softmax with
// DPP reductions; both QK^T and P*V on WMMA.
__global__ __launch_bounds__(128)
void attn_kernel(const unsigned short* __restrict__ qg, const unsigned short* __restrict__ kg,
                 const unsigned short* __restrict__ vg, const float* __restrict__ validf,
                 unsigned short* __restrict__ og, int L, int qgroups) {
  __shared__ unsigned short Ks[128 * 64];        // [key][dk]
  __shared__ unsigned short Vt[64 * 128];        // transposed: [dv][key]
  __shared__ float maskadd[128];
  __shared__ unsigned short Pl[4][16 * 32];      // per-wave P scratch (A-frag staging)

  const int tid  = threadIdx.x;
  const int lane = tid & 31, lr = lane & 15, hi = lane >> 4;
  const int w = tid >> 5;
  const int qgi = blockIdx.x % qgroups;
  const int bh  = blockIdx.x / qgroups;
  const int b = bh >> 3, h = bh & 7;
  const int qt = qgi * 4 + w;

  // Q fragments for this wave's 16 query rows (dk 0..31 and 32..63)
  FragBF aq0, aq1;
  {
    int qrow = qt * 16 + lr; if (qrow > L - 1) qrow = L - 1;
    const unsigned short* qp = qg + ((size_t)(b * L + qrow)) * DMODEL + h * DKH;
    aq0.q[0] = *(const uint4*)(qp + hi * 16);       aq0.q[1] = *(const uint4*)(qp + hi * 16 + 8);
    aq1.q[0] = *(const uint4*)(qp + 32 + hi * 16);  aq1.q[1] = *(const uint4*)(qp + 32 + hi * 16 + 8);
  }

  float m[8], s[8];
  v8f o0 = {}, o1 = {}, o2 = {}, o3 = {};
#pragma unroll
  for (int r = 0; r < 8; ++r) { m[r] = -1e30f; s[r] = 0.0f; }

  const int nchunk = (L + 127) / 128;
  for (int ck = 0; ck < nchunk; ++ck) {
    const int kc0 = ck * 128;
    __syncthreads();
    { // stage K rows via async DMA + mask addend (stale pad rows are masked out)
      const int row = tid;                       // 0..127
      const bool ok = (kc0 + row) < L;
      if (ok) {
        const unsigned short* kp = kg + ((size_t)(b * L + kc0 + row)) * DMODEL + h * DKH;
#pragma unroll
        for (int c = 0; c < 8; ++c)
          async_copy_b128((unsigned)(size_t)(Ks + row * 64 + c * 8), kp + c * 8);
      }
      maskadd[row] = (ok && validf[b * L + kc0 + row] > 0.5f) ? 0.0f : -1e9f;
    }
    // stage V transposed: b128 global loads + b16 LDS scatter
#pragma unroll
    for (int p = 0; p < 8; ++p) {
      const int idx = tid + p * 128;             // 0..1023
      const int j = idx >> 3, c = (idx & 7) * 8; // key, dv chunk
      U4H d; d.q = make_uint4(0, 0, 0, 0);
      if (kc0 + j < L)
        d.q = *(const uint4*)(vg + ((size_t)(b * L + kc0 + j)) * DMODEL + h * DKH + c);
#pragma unroll
      for (int u = 0; u < 8; ++u) Vt[(c + u) * 128 + j] = d.h[u];
    }
    // prefetch next chunk of K/V
    if (kc0 + 128 < L) {
      __builtin_prefetch(kg + ((size_t)(b * L + kc0 + 128 + tid)) * DMODEL + h * DKH, 0, 0);
      __builtin_prefetch(vg + ((size_t)(b * L + kc0 + 128 + tid)) * DMODEL + h * DKH, 0, 0);
    }
    wait_async0();                 // K DMA landed in LDS
    __syncthreads();

#pragma unroll 1
    for (int ks = 0; ks < 4; ++ks) {
      const int kloc = ks * 32;
      // ---- scores: two 16x16 S tiles over 32 keys, contraction over dk=64
      v8f sc0 = {}, sc1 = {};
      FragBF bk;
      const unsigned short* kr0 = Ks + (kloc + lr) * 64;
      bk.q[0] = *(const uint4*)(kr0 + hi * 16);      bk.q[1] = *(const uint4*)(kr0 + hi * 16 + 8);
      sc0 = wmma_bf16(aq0, bk, sc0);
      bk.q[0] = *(const uint4*)(kr0 + 32 + hi * 16); bk.q[1] = *(const uint4*)(kr0 + 32 + hi * 16 + 8);
      sc0 = wmma_bf16(aq1, bk, sc0);
      const unsigned short* kr1 = Ks + (kloc + 16 + lr) * 64;
      bk.q[0] = *(const uint4*)(kr1 + hi * 16);      bk.q[1] = *(const uint4*)(kr1 + hi * 16 + 8);
      sc1 = wmma_bf16(aq0, bk, sc1);
      bk.q[0] = *(const uint4*)(kr1 + 32 + hi * 16); bk.q[1] = *(const uint4*)(kr1 + 32 + hi * 16 + 8);
      sc1 = wmma_bf16(aq1, bk, sc1);

      const float ma0 = maskadd[kloc + lr], ma1 = maskadd[kloc + 16 + lr];
      unsigned short* Pw = Pl[w];
#pragma unroll
      for (int r = 0; r < 8; ++r) {
        float v0 = sc0[r] * 0.125f + ma0;          // 1/sqrt(64)
        float v1 = sc1[r] * 0.125f + ma1;
        float mx = redmax16(fmaxf(v0, v1));        // DPP butterfly, no LDS
        const float nm = fmaxf(m[r], mx);
        const float alpha = __expf(m[r] - nm);
        m[r] = nm;
        const float p0 = __expf(v0 - nm), p1 = __expf(v1 - nm);
        const float ps = redsum16(p0 + p1);        // DPP butterfly, no LDS
        s[r] = s[r] * alpha + ps;
        o0[r] *= alpha; o1[r] *= alpha; o2[r] *= alpha; o3[r] *= alpha;
        Pw[(r + 8 * hi) * 32 + lr]      = f2bf(p0);
        Pw[(r + 8 * hi) * 32 + 16 + lr] = f2bf(p1);
      }
      asm volatile("s_wait_dscnt 0" ::: "memory");   // LDS RAW before A-frag readback
      FragBF ap, bv;
      ap.q[0] = *(const uint4*)(Pw + lr * 32 + hi * 16);
      ap.q[1] = *(const uint4*)(Pw + lr * 32 + hi * 16 + 8);
      const unsigned short* vr;
      vr = Vt + (0 * 16 + lr) * 128 + kloc + hi * 16;
      bv.q[0] = *(const uint4*)vr; bv.q[1] = *(const uint4*)(vr + 8);
      o0 = wmma_bf16(ap, bv, o0);
      vr = Vt + (1 * 16 + lr) * 128 + kloc + hi * 16;
      bv.q[0] = *(const uint4*)vr; bv.q[1] = *(const uint4*)(vr + 8);
      o1 = wmma_bf16(ap, bv, o1);
      vr = Vt + (2 * 16 + lr) * 128 + kloc + hi * 16;
      bv.q[0] = *(const uint4*)vr; bv.q[1] = *(const uint4*)(vr + 8);
      o2 = wmma_bf16(ap, bv, o2);
      vr = Vt + (3 * 16 + lr) * 128 + kloc + hi * 16;
      bv.q[0] = *(const uint4*)vr; bv.q[1] = *(const uint4*)(vr + 8);
      o3 = wmma_bf16(ap, bv, o3);
    }
  }

#pragma unroll
  for (int r = 0; r < 8; ++r) {
    const float inv = 1.0f / s[r];
    const int row = qt * 16 + r + 8 * hi;
    if (row < L) {
      const size_t base = ((size_t)(b * L + row)) * DMODEL + h * DKH + lr;
      og[base + 0]  = f2bf(o0[r] * inv);
      og[base + 16] = f2bf(o1[r] * inv);
      og[base + 32] = f2bf(o2[r] * inv);
      og[base + 48] = f2bf(o3[r] * inv);
    }
  }
}

__global__ void writeout_kernel(const float* __restrict__ x, const float* __restrict__ validf,
                                float* __restrict__ out, int BL) {
  const int n = BL * DMODEL;
  for (int i = blockIdx.x * blockDim.x + threadIdx.x; i < n + BL; i += gridDim.x * blockDim.x) {
    out[i] = (i < n) ? x[i] : validf[i - n];
  }
}

// ------------------------------------------------------------------- host
extern "C" void kernel_launch(void* const* d_in, const int* in_sizes, int n_in,
                              void* d_out, int out_size, void* d_ws, size_t ws_size,
                              hipStream_t stream) {
  const int*   poi_type = (const int*)d_in[0];
  const float* loc_emb  = (const float*)d_in[1];
  const int*   npl      = (const int*)d_in[2];
  const int*   ttn      = (const int*)d_in[3];
  const float* emb      = (const float*)d_in[4];
  const float* Wq = (const float*)d_in[5];
  const float* Wk = (const float*)d_in[6];
  const float* Wv = (const float*)d_in[7];
  const float* Wo = (const float*)d_in[8];
  const float* bo = (const float*)d_in[9];
  const float* ln1g = (const float*)d_in[10];
  const float* ln1b = (const float*)d_in[11];
  const float* W1 = (const float*)d_in[12];
  const float* b1 = (const float*)d_in[13];
  const float* W2 = (const float*)d_in[14];
  const float* b2 = (const float*)d_in[15];
  const float* ln2g = (const float*)d_in[16];
  const float* ln2b = (const float*)d_in[17];
  const float* lnfg = (const float*)d_in[18];
  const float* lnfb = (const float*)d_in[19];
  (void)in_sizes; (void)n_in; (void)ws_size;

  const int L = out_size / (BATCH * (DMODEL + 1));   // out = [B,L,D] + mask [B,L]
  if (L <= 0) return;
  const int BL = BATCH * L;

  // --- workspace carve (256B aligned regions)
  char* base = (char*)d_ws;
  size_t off = 0;
  auto carve = [&](size_t bytes) -> char* {
    char* p = base + off; off = (off + bytes + 255) & ~(size_t)255; return p;
  };
  int*   poi_idx = (int*)carve(sizeof(int) * BL);
  int*   tok_idx = (int*)carve(sizeof(int) * BL);
  float* validf  = (float*)carve(sizeof(float) * BL);
  const size_t szD2 = (size_t)NLAYERS * DMODEL * DMODEL;   // 1,048,576
  const size_t szDI = (size_t)NLAYERS * DMODEL * DINNER;   // 4,194,304
  unsigned short* WqB = (unsigned short*)carve(2 * szD2);
  unsigned short* WkB = (unsigned short*)carve(2 * szD2);
  unsigned short* WvB = (unsigned short*)carve(2 * szD2);
  unsigned short* WoB = (unsigned short*)carve(2 * szD2);
  unsigned short* W1B = (unsigned short*)carve(2 * szDI);
  unsigned short* W2B = (unsigned short*)carve(2 * szDI);
  float*          x   = (float*)carve(sizeof(float) * (size_t)BL * DMODEL);
  unsigned short* xb  = (unsigned short*)carve(2ull * BL * DMODEL);
  unsigned short* qb  = (unsigned short*)carve(2ull * BL * DMODEL);
  unsigned short* kb  = (unsigned short*)carve(2ull * BL * DMODEL);
  unsigned short* vb  = (unsigned short*)carve(2ull * BL * DMODEL);
  unsigned short* ob  = (unsigned short*)carve(2ull * BL * DMODEL);
  float*          tmp = (float*)carve(sizeof(float) * (size_t)BL * DMODEL);
  unsigned short* hb  = (unsigned short*)carve(2ull * BL * DINNER);

  // --- weight conversion f32 -> bf16
  auto cvt = [&](const float* src, unsigned short* dst, size_t n) {
    cvt_bf16_kernel<<<1024, 256, 0, stream>>>(src, dst, (int)n);
  };
  cvt(Wq, WqB, szD2); cvt(Wk, WkB, szD2); cvt(Wv, WvB, szD2); cvt(Wo, WoB, szD2);
  cvt(W1, W1B, szDI); cvt(W2, W2B, szDI);

  // --- ragged pack + embedding + final LN
  pack_kernel<<<1, 64, 0, stream>>>(npl, ttn, poi_idx, tok_idx, validf, L);
  embed_ln_kernel<<<BL, 256, 0, stream>>>(poi_type, loc_emb, emb, poi_idx, tok_idx,
                                          validf, lnfg, lnfb, x, xb, L);

  const int gm = (BL + 127) / 128;
  const dim3 gD(DMODEL / 64, gm);    // N=512 tiles
  const dim3 gI(DINNER / 64, gm);    // N=2048 tiles
  const int ntile = (L + 15) / 16;
  const int qgroups = (ntile + 3) / 4;
  const int attn_grid = BATCH * NHEAD * qgroups;

  for (int l = 0; l < NLAYERS; ++l) {
    const unsigned short* wq = WqB + (size_t)l * DMODEL * DMODEL;
    const unsigned short* wk = WkB + (size_t)l * DMODEL * DMODEL;
    const unsigned short* wv = WvB + (size_t)l * DMODEL * DMODEL;
    const unsigned short* wo = WoB + (size_t)l * DMODEL * DMODEL;
    const unsigned short* w1 = W1B + (size_t)l * DMODEL * DINNER;
    const unsigned short* w2 = W2B + (size_t)l * DINNER * DMODEL;

    // QKV projections (bf16 out, no bias)
    gemm_kernel<1, 0, 0><<<gD, 256, 0, stream>>>(xb, wq, nullptr, nullptr, qb, BL, DMODEL, DMODEL);
    gemm_kernel<1, 0, 0><<<gD, 256, 0, stream>>>(xb, wk, nullptr, nullptr, kb, BL, DMODEL, DMODEL);
    gemm_kernel<1, 0, 0><<<gD, 256, 0, stream>>>(xb, wv, nullptr, nullptr, vb, BL, DMODEL, DMODEL);
    // attention
    attn_kernel<<<attn_grid, 128, 0, stream>>>(qb, kb, vb, validf, ob, L, qgroups);
    // output projection (+bias) -> tmp, then residual + LN1
    gemm_kernel<0, 0, 1><<<gD, 256, 0, stream>>>(ob, wo, bo + (size_t)l * DMODEL,
                                                 tmp, nullptr, BL, DMODEL, DMODEL);
    resln_kernel<<<BL, 256, 0, stream>>>(x, tmp, ln1g + (size_t)l * DMODEL,
                                         ln1b + (size_t)l * DMODEL, xb);
    // FFN: relu(x W1 + b1) W2 + b2, residual + LN2
    gemm_kernel<1, 1, 1><<<gI, 256, 0, stream>>>(xb, w1, b1 + (size_t)l * DINNER,
                                                 nullptr, hb, BL, DINNER, DMODEL);
    gemm_kernel<0, 0, 1><<<gD, 256, 0, stream>>>(hb, w2, b2 + (size_t)l * DMODEL,
                                                 tmp, nullptr, BL, DMODEL, DINNER);
    resln_kernel<<<BL, 256, 0, stream>>>(x, tmp, ln2g + (size_t)l * DMODEL,
                                         ln2b + (size_t)l * DMODEL, xb);
  }

  const int nout = BL * (DMODEL + 1);
  writeout_kernel<<<(nout + 255) / 256, 256, 0, stream>>>(x, validf, (float*)d_out, BL);
}
